// SelfAttention_5566277615693
// MI455X (gfx1250) — compile-verified, matrix operand order checked by reading, not run on hardware
//
#include <hip/hip_runtime.h>
#include <hip/hip_bf16.h>

// ---------------------------------------------------------------------------
// Self-attention for B=4, S=2048, D=1024 on gfx1250 (MI455X).
// Heavy math: v_wmma_f32_16x16x32_bf16 (16 WMMA per wave per K-step).
// Data movement: global_load_async_to_lds_b128 (ASYNCcnt), no VGPR staging.
// ---------------------------------------------------------------------------

typedef __bf16 bf16_t;
typedef __attribute__((ext_vector_type(16))) __bf16 v16bf;
typedef __attribute__((ext_vector_type(8)))  float  v8f;

#define ATT_B 4
#define ATT_S 2048
#define ATT_D 1024
#define BM 128
#define BN 256
#define BK 32
#define LDS_STRIDE (BK + 8)   // 40 bf16 = 80B rows -> conflict-free b128 frag loads

// Async global->LDS copy of 16 bytes per lane (CDNA5, ASYNCcnt-tracked).
// VDST = VGPR holding LDS byte address; VADDR = 64-bit global address.
__device__ __forceinline__ void async_copy_b128(void* lds_dst, const void* gsrc) {
  uint32_t lds = (uint32_t)(uintptr_t)lds_dst;   // low 32 bits of generic ptr = LDS offset
  asm volatile("global_load_async_to_lds_b128 %0, %1, off"
               :
               : "v"(lds), "v"(gsrc)
               : "memory");
}

__device__ __forceinline__ void wait_asynccnt0() {
  asm volatile("s_wait_asynccnt 0x0" ::: "memory");
}

// ---------------------------------------------------------------------------
// f32 -> bf16 elementwise conversion (vectorized x4)
// ---------------------------------------------------------------------------
__global__ __launch_bounds__(256) void f32_to_bf16_kernel(
    const float* __restrict__ in, bf16_t* __restrict__ out, long n) {
  long i = ((long)blockIdx.x * blockDim.x + threadIdx.x) * 4;
  if (i + 3 < n) {
    float4 f = *(const float4*)(in + i);
    out[i + 0] = (bf16_t)f.x;
    out[i + 1] = (bf16_t)f.y;
    out[i + 2] = (bf16_t)f.z;
    out[i + 3] = (bf16_t)f.w;
  }
}

// ---------------------------------------------------------------------------
// NT GEMM: C[m,n] = scale * sum_k A[m,k]*B[n,k] (+ bias[n])
//   A: [M,K] row-major bf16 (lda), B: [N,K] row-major bf16 (ldb)
//   grid.x = M/128, grid.y = N/256, grid.z = batch (strides in elements)
//   TRANSOUT: C[n*ldc + m] (used to materialize V^T), else C[m*ldc + n].
// Block = 256 threads = 8 waves (2 in M x 4 in N); wave tile 64x64 =
// 4x4 fragments of 16x16 -> 16 WMMAs per K-step of 32 against
// 16 ds_load_b128 (1:1 wmma:ds ratio). Global->LDS staging is
// asynchronous (double-buffered): one s_wait_asynccnt + one barrier/K-step.
// ---------------------------------------------------------------------------
template <typename OutT, bool BIAS, bool TRANSOUT>
__global__ __launch_bounds__(256) void gemm_nt_wmma(
    const bf16_t* __restrict__ A, long strideA, int lda,
    const bf16_t* __restrict__ Bm, long strideB, int ldb,
    OutT* __restrict__ C, long strideC, int ldc,
    const float* __restrict__ bias, int K, float scale) {
  __shared__ bf16_t As[2][BM][LDS_STRIDE];   // 2*128*40*2 = 20 KB
  __shared__ bf16_t Bs[2][BN][LDS_STRIDE];   // 2*256*40*2 = 40 KB

  A  += (long)blockIdx.z * strideA;
  Bm += (long)blockIdx.z * strideB;
  C  += (long)blockIdx.z * strideC;

  const int tid  = threadIdx.x;
  const int lane = tid & 31;
  const int wid  = tid >> 5;
  const int wm   = (wid >> 2) * 64;   // wave M offset within block tile (2 waves)
  const int wn   = (wid & 3) * 64;    // wave N offset within block tile (4 waves)
  const int half = lane >> 4;         // which 16-lane half
  const int lrow = lane & 15;
  const int koff = half * 8;          // ISA 16-bit A/B striping: K {0-7,16-23} / {8-15,24-31}

  const long blockM = (long)blockIdx.x * BM;
  const long blockN = (long)blockIdx.y * BN;
  const bf16_t* Ag = A  + blockM * lda;
  const bf16_t* Bg = Bm + blockN * ldb;

  v8f acc[4][4];
  const v8f vzero = {0.f, 0.f, 0.f, 0.f, 0.f, 0.f, 0.f, 0.f};
#pragma unroll
  for (int i = 0; i < 4; ++i)
#pragma unroll
    for (int j = 0; j < 4; ++j) acc[i][j] = vzero;

  // A tile: 128x32 bf16 = 512 x 16B chunks (2/thread);
  // B tile: 256x32 bf16 = 1024 x 16B chunks (4/thread). All async, no waits.
  auto load_stage_async = [&](int buf, int k0) {
#pragma unroll
    for (int t = 0; t < 2; ++t) {
      int c   = tid + t * 256;       // 0..511
      int row = c >> 2;
      int col = (c & 3) * 8;
      async_copy_b128(&As[buf][row][col], &Ag[(long)row * lda + k0 + col]);
    }
#pragma unroll
    for (int t = 0; t < 4; ++t) {
      int c   = tid + t * 256;       // 0..1023
      int row = c >> 2;
      int col = (c & 3) * 8;
      async_copy_b128(&Bs[buf][row][col], &Bg[(long)row * ldb + k0 + col]);
    }
  };

  load_stage_async(0, 0);
  wait_asynccnt0();
  __syncthreads();

  const int KT = K / BK;
  for (int kt = 0; kt < KT; ++kt) {
    const int buf = kt & 1;
    if (kt + 1 < KT) load_stage_async(buf ^ 1, (kt + 1) * BK);  // background prefetch

    v16bf a[4], b[4];
#pragma unroll
    for (int i = 0; i < 4; ++i) {
      const bf16_t* p = &As[buf][wm + i * 16 + lrow][koff];
      union { uint4 u[2]; v16bf v; } t;
      t.u[0] = *(const uint4*)p;
      t.u[1] = *(const uint4*)(p + 16);
      a[i] = t.v;
    }
#pragma unroll
    for (int j = 0; j < 4; ++j) {
      const bf16_t* p = &Bs[buf][wn + j * 16 + lrow][koff];
      union { uint4 u[2]; v16bf v; } t;
      t.u[0] = *(const uint4*)p;
      t.u[1] = *(const uint4*)(p + 16);
      b[j] = t.v;
    }
#pragma unroll
    for (int i = 0; i < 4; ++i)
#pragma unroll
      for (int j = 0; j < 4; ++j)
        acc[i][j] = __builtin_amdgcn_wmma_f32_16x16x32_bf16(
            false, a[i], false, b[j], (short)0, acc[i][j], false, false);

    wait_asynccnt0();       // own async writes to buf^1 landed in LDS
    __syncthreads();        // everyone's writes visible; safe to flip buffers
  }

  // Epilogue. C/D layout: VGPR r -> M = r (lanes 0-15) / r+8 (lanes 16-31); N = lane&15.
#pragma unroll
  for (int i = 0; i < 4; ++i) {
#pragma unroll
    for (int j = 0; j < 4; ++j) {
      const long n  = blockN + wn + j * 16 + lrow;
      const long m0 = blockM + wm + i * 16 + half * 8;
      float bv = 0.f;
      if (BIAS) bv = bias[n];
#pragma unroll
      for (int r = 0; r < 8; ++r) {
        float v = acc[i][j][r] * scale + bv;
        if (TRANSOUT)
          C[n * (long)ldc + m0 + r] = (OutT)v;
        else
          C[(m0 + r) * (long)ldc + n] = (OutT)v;
      }
    }
  }
}

// ---------------------------------------------------------------------------
// In-place row softmax over bf16 scores (f32 math). One block per row of 2048.
// ---------------------------------------------------------------------------
__global__ __launch_bounds__(256) void softmax_rows_bf16(bf16_t* __restrict__ P, int ncols) {
  __shared__ float red[256];
  bf16_t* row = P + (long)blockIdx.x * ncols;
  const int tid = threadIdx.x;

  union { uint4 u; bf16_t h[8]; } ld;
  ld.u = *(const uint4*)(row + tid * 8);

  float v[8];
  float m = -3.0e38f;
#pragma unroll
  for (int r = 0; r < 8; ++r) {
    v[r] = (float)ld.h[r];
    m = fmaxf(m, v[r]);
  }
  red[tid] = m;
  __syncthreads();
  for (int s = 128; s > 0; s >>= 1) {
    if (tid < s) red[tid] = fmaxf(red[tid], red[tid + s]);
    __syncthreads();
  }
  m = red[0];
  __syncthreads();

  float sum = 0.f;
#pragma unroll
  for (int r = 0; r < 8; ++r) {
    v[r] = __expf(v[r] - m);
    sum += v[r];
  }
  red[tid] = sum;
  __syncthreads();
  for (int s = 128; s > 0; s >>= 1) {
    if (tid < s) red[tid] += red[tid + s];
    __syncthreads();
  }
  const float inv = 1.f / red[0];

  union { uint4 u; bf16_t h[8]; } st;
#pragma unroll
  for (int r = 0; r < 8; ++r) st.h[r] = (bf16_t)(v[r] * inv);
  *(uint4*)(row + tid * 8) = st.u;
}

// ---------------------------------------------------------------------------
// Host-side orchestration (graph-capture safe: only kernel launches on stream)
// ---------------------------------------------------------------------------
extern "C" void kernel_launch(void* const* d_in, const int* in_sizes, int n_in,
                              void* d_out, int out_size, void* d_ws, size_t ws_size,
                              hipStream_t stream) {
  (void)in_sizes; (void)n_in; (void)out_size; (void)ws_size;

  const float* x  = (const float*)d_in[0];
  const float* Wq = (const float*)d_in[1];
  const float* bq = (const float*)d_in[2];
  const float* Wk = (const float*)d_in[3];
  const float* bk = (const float*)d_in[4];
  const float* Wv = (const float*)d_in[5];
  const float* bv = (const float*)d_in[6];
  float* out = (float*)d_out;

  const long BS = (long)ATT_B * ATT_S;        // 8192 flattened rows
  const long Dd = ATT_D;                      // 1024
  const long S  = ATT_S;                      // 2048

  // Workspace carve-up (bytes): x_bf16 | Wq | Wk | Wv | Q | K | Vt | scores
  char* p = (char*)d_ws;
  bf16_t* xb  = (bf16_t*)p; p += BS * Dd * 2;        // 16 MB
  bf16_t* wqb = (bf16_t*)p; p += Dd * Dd * 2;        //  2 MB
  bf16_t* wkb = (bf16_t*)p; p += Dd * Dd * 2;        //  2 MB
  bf16_t* wvb = (bf16_t*)p; p += Dd * Dd * 2;        //  2 MB
  bf16_t* Qb  = (bf16_t*)p; p += BS * Dd * 2;        // 16 MB
  bf16_t* Kb  = (bf16_t*)p; p += BS * Dd * 2;        // 16 MB
  bf16_t* Vtb = (bf16_t*)p; p += Dd * BS * 2;        // 16 MB  (V^T: [D][B*S])
  bf16_t* Sc  = (bf16_t*)p;                          // 32 MB  (scores/probs bf16)

  // 1) convert x and weights to bf16
  f32_to_bf16_kernel<<<dim3((unsigned)(BS * Dd / 1024)), 256, 0, stream>>>(x, xb, BS * Dd);
  f32_to_bf16_kernel<<<dim3((unsigned)(Dd * Dd / 1024)), 256, 0, stream>>>(Wq, wqb, Dd * Dd);
  f32_to_bf16_kernel<<<dim3((unsigned)(Dd * Dd / 1024)), 256, 0, stream>>>(Wk, wkb, Dd * Dd);
  f32_to_bf16_kernel<<<dim3((unsigned)(Dd * Dd / 1024)), 256, 0, stream>>>(Wv, wvb, Dd * Dd);

  // 2) projections (NT: y = x @ W^T + b). M=8192, N=1024, K=1024.
  dim3 gProj((unsigned)(BS / BM), (unsigned)(Dd / BN), 1);
  gemm_nt_wmma<bf16_t, true, false><<<gProj, 256, 0, stream>>>(
      xb, 0, (int)Dd, wqb, 0, (int)Dd, Qb, 0, (int)Dd, bq, (int)Dd, 1.0f);
  gemm_nt_wmma<bf16_t, true, false><<<gProj, 256, 0, stream>>>(
      xb, 0, (int)Dd, wkb, 0, (int)Dd, Kb, 0, (int)Dd, bk, (int)Dd, 1.0f);
  // V stored transposed: Vt[n][m_global], ldc = 8192, so PV is also NT.
  gemm_nt_wmma<bf16_t, true, true><<<gProj, 256, 0, stream>>>(
      xb, 0, (int)Dd, wvb, 0, (int)Dd, Vtb, 0, (int)BS, bv, (int)Dd, 1.0f);

  // 3) scores = scale * Q @ K^T, per batch. M=N=2048, K=1024. scale = 1/sqrt(1024).
  dim3 gScore((unsigned)(S / BM), (unsigned)(S / BN), ATT_B);
  gemm_nt_wmma<bf16_t, false, false><<<gScore, 256, 0, stream>>>(
      Qb, S * Dd, (int)Dd, Kb, S * Dd, (int)Dd, Sc, S * S, (int)S,
      nullptr, (int)Dd, 0.03125f);

  // 4) row softmax in place (8192 rows of 2048)
  softmax_rows_bf16<<<dim3((unsigned)BS), 256, 0, stream>>>(Sc, (int)S);

  // 5) O = P @ V  ==  P @ (V^T)^T, NT against Vt (batch b = columns b*2048..).
  dim3 gPV((unsigned)(S / BM), (unsigned)(Dd / BN), ATT_B);
  gemm_nt_wmma<float, false, false><<<gPV, 256, 0, stream>>>(
      Sc, S * S, (int)S, Vtb, S /*col offset per batch*/, (int)BS,
      out, S * Dd, (int)Dd, nullptr, (int)S, 1.0f);
}
